// RNNDecoder_26903675142687
// MI455X (gfx1250) — matrix operand order repaired and checked by reading, not compiled, not run on hardware
//
#include <hip/hip_runtime.h>

// ---------------------------------------------------------------------------
// RNN greedy decoder for MI455X (gfx1250, wave32, WMMA).
// B=64, IN=2048, H=1024, L=2, V=32000, T=60.
// bf16 weights (L2-resident across the 60 steps) + v_wmma_f32_16x16x32_bf16.
// Round 3: explicit double-buffered K-loop (loads one chunk ahead of the WMMA
// chain) and the LSTM cell fused into the gate GEMM epilogue (i/f/g/o strips
// computed by the same wave; h ping-pong buffered, c updated in place).
// ---------------------------------------------------------------------------

typedef __attribute__((ext_vector_type(16))) __bf16 v16bf;
typedef __attribute__((ext_vector_type(8)))  float  v8f;

#define RD_B   64
#define RD_IN  2048
#define RD_H   1024
#define RD_G   4096
#define RD_V   32000
#define RD_T   60

__device__ __forceinline__ unsigned short f2bf(float f) {
    unsigned int u = __float_as_uint(f);
    unsigned int r = u + 0x7FFFu + ((u >> 16) & 1u);   // round to nearest even
    return (unsigned short)(r >> 16);
}

__device__ __forceinline__ float sigm(float x) { return 1.0f / (1.0f + __expf(-x)); }

union Frag16 {            // 16 bf16 values = one WMMA A/B fragment per lane
    v16bf v;
    uint4 q[2];
};

// A fragment (16x32 bf16, row-major, lda=K):
//   lane l<16: M=l, K=0..15;  lane l>=16: M=l-16, K=16..31.
#define LOAD_A(dst, rowPtr, k0, half)                                   \
    do {                                                                \
        (dst).q[0] = *(const uint4*)((rowPtr) + (k0) + (half) * 16);    \
        (dst).q[1] = *(const uint4*)((rowPtr) + (k0) + 8 + (half) * 16);\
    } while (0)

// B fragment (32x16 bf16): B[k][n]=W[n][k]; lane: N=l&15, K=(l>>4)*16+0..15.
// rowPtr is pre-offset by (col+n)*K + half*16.
#define LOAD_B(dst, rowPtr, k0)                                         \
    do {                                                                \
        (dst).q[0] = *(const uint4*)((rowPtr) + (k0));                  \
        (dst).q[1] = *(const uint4*)((rowPtr) + (k0) + 8);              \
    } while (0)

#define WMMA_BF16(accv, av, bv)                                         \
    (accv) = __builtin_amdgcn_wmma_f32_16x16x32_bf16(                   \
        false, (av), false, (bv), (short)0, (accv), false, false)

// ---------------------------------------------------------------------------
// Skinny GEMM:  Out[64,N] = A[64,K] * W[N,K]^T + bias ; optional ReLU;
// f32 and/or bf16 output.  256 threads = 8 waves; wave w: rows (w&3)*16,
// cols blockIdx.x*128 + (w>>2)*64 (4 accumulators of 16x16).
// K-loop double-buffered: chunk k+32 is loaded before chunk k's WMMAs.
// ---------------------------------------------------------------------------
template<bool RELU, bool OUTF, bool OUTBF>
__global__ __launch_bounds__(256)
void rd_gemm(const unsigned short* __restrict__ A,
             const unsigned short* __restrict__ W,
             const float* __restrict__ bias,
             float* __restrict__ outF,
             unsigned short* __restrict__ outBf,
             int N, int K, int ldo) {
    const int w       = threadIdx.x >> 5;
    const int lane    = threadIdx.x & 31;
    const int row0    = (w & 3) * 16;
    const int colBase = blockIdx.x * 128 + (w >> 2) * 64;
    const int m       = lane & 15;
    const int half    = lane >> 4;

    const unsigned short* aRow = A + (size_t)(row0 + m) * K;
    const unsigned short* wRow[4];
    #pragma unroll
    for (int j = 0; j < 4; ++j)
        wRow[j] = W + (size_t)(colBase + j * 16 + m) * K + half * 16;

    v8f acc[4];
    #pragma unroll
    for (int j = 0; j < 4; ++j) {
        const float bv = bias[colBase + j * 16 + m];
        #pragma unroll
        for (int r = 0; r < 8; ++r) acc[j][r] = bv;
    }

    // prologue: load chunk 0
    Frag16 aC, bC[4];
    LOAD_A(aC, aRow, 0, half);
    #pragma unroll
    for (int j = 0; j < 4; ++j) LOAD_B(bC[j], wRow[j], 0);

    for (int k0 = 0; k0 + 32 < K; k0 += 32) {
        const int kn = k0 + 32;
        Frag16 aN, bN[4];
        LOAD_A(aN, aRow, kn, half);
        #pragma unroll
        for (int j = 0; j < 4; ++j) LOAD_B(bN[j], wRow[j], kn);
        if (kn + 32 < K) {                      // global_prefetch_b8
            __builtin_prefetch(aRow + kn + 32, 0, 3);
            __builtin_prefetch(wRow[0] + kn + 32, 0, 3);
        }
        #pragma unroll
        for (int j = 0; j < 4; ++j) WMMA_BF16(acc[j], aC.v, bC[j].v);
        aC = aN;
        #pragma unroll
        for (int j = 0; j < 4; ++j) bC[j] = bN[j];
    }
    #pragma unroll
    for (int j = 0; j < 4; ++j) WMMA_BF16(acc[j], aC.v, bC[j].v);

    if (RELU) {
        #pragma unroll
        for (int j = 0; j < 4; ++j)
            #pragma unroll
            for (int r = 0; r < 8; ++r) acc[j][r] = fmaxf(acc[j][r], 0.0f);
    }

    // D layout: lane holds col n=lane&15; VGPR r holds row M = half*8 + r.
    #pragma unroll
    for (int j = 0; j < 4; ++j) {
        const int col = colBase + j * 16 + m;
        #pragma unroll
        for (int r = 0; r < 8; ++r) {
            const size_t idx = (size_t)(row0 + half * 8 + r) * (size_t)ldo + col;
            if (OUTF)  outF[idx]  = acc[j][r];
            if (OUTBF) outBf[idx] = f2bf(acc[j][r]);
        }
    }
}

// ---------------------------------------------------------------------------
// Fused LSTM layer: gates = inp @ W_ih^T + h_prev @ W_hh^T + b_ih + b_hh,
// then the cell nonlinearity, all in one kernel.
// Block owns 32 h-columns; wave w: rows (w&3)*16, h-cols (w>>2)*16.
// acc[g] is the (rows x 16) tile of gate g at column g*1024 + hcol.
// After the K-loop each lane holds i/f/g/o for 8 (batch,h) pairs -> cell math.
// h_out must differ from h_prev (ping-pong); c updated in place (one owner).
// ---------------------------------------------------------------------------
__global__ __launch_bounds__(256)
void rd_lstm_layer(const unsigned short* __restrict__ inp,     // [64,1024] bf16
                   const unsigned short* __restrict__ h_prev,  // [64,1024] bf16
                   const unsigned short* __restrict__ W_ih,    // [4096,1024] bf16
                   const unsigned short* __restrict__ W_hh,    // [4096,1024] bf16
                   const float* __restrict__ b_ih,             // [4096]
                   const float* __restrict__ b_hh,             // [4096]
                   float* __restrict__ c_st,                   // [64,1024] f32
                   unsigned short* __restrict__ h_out) {       // [64,1024] bf16
    const int w     = threadIdx.x >> 5;
    const int lane  = threadIdx.x & 31;
    const int row0  = (w & 3) * 16;
    const int hcol0 = blockIdx.x * 32 + (w >> 2) * 16;
    const int m     = lane & 15;
    const int half  = lane >> 4;
    const int K     = RD_H;

    const unsigned short* aRow  = inp    + (size_t)(row0 + m) * K;
    const unsigned short* a2Row = h_prev + (size_t)(row0 + m) * K;
    const unsigned short* wRow[4];   // W_ih rows for gates i,f,g,o
    const unsigned short* w2Row[4];  // W_hh rows
    #pragma unroll
    for (int g = 0; g < 4; ++g) {
        const size_t n = (size_t)(g * RD_H + hcol0 + m);
        wRow[g]  = W_ih + n * K + half * 16;
        w2Row[g] = W_hh + n * K + half * 16;
    }

    v8f acc[4];
    #pragma unroll
    for (int g = 0; g < 4; ++g) {
        const int n = g * RD_H + hcol0 + m;
        const float bv = b_ih[n] + b_hh[n];
        #pragma unroll
        for (int r = 0; r < 8; ++r) acc[g][r] = bv;
    }

    // prologue
    Frag16 aC, a2C, bC[4], b2C[4];
    LOAD_A(aC,  aRow,  0, half);
    LOAD_A(a2C, a2Row, 0, half);
    #pragma unroll
    for (int g = 0; g < 4; ++g) { LOAD_B(bC[g], wRow[g], 0); LOAD_B(b2C[g], w2Row[g], 0); }

    for (int k0 = 0; k0 + 32 < K; k0 += 32) {
        const int kn = k0 + 32;
        Frag16 aN, a2N, bN[4], b2N[4];
        LOAD_A(aN,  aRow,  kn, half);
        LOAD_A(a2N, a2Row, kn, half);
        #pragma unroll
        for (int g = 0; g < 4; ++g) { LOAD_B(bN[g], wRow[g], kn); LOAD_B(b2N[g], w2Row[g], kn); }
        if (kn + 32 < K) {
            __builtin_prefetch(wRow[0] + kn + 32, 0, 3);
            __builtin_prefetch(w2Row[0] + kn + 32, 0, 3);
        }
        #pragma unroll
        for (int g = 0; g < 4; ++g) {
            WMMA_BF16(acc[g], aC.v,  bC[g].v);
            WMMA_BF16(acc[g], a2C.v, b2C[g].v);
        }
        aC = aN; a2C = a2N;
        #pragma unroll
        for (int g = 0; g < 4; ++g) { bC[g] = bN[g]; b2C[g] = b2N[g]; }
    }
    #pragma unroll
    for (int g = 0; g < 4; ++g) {
        WMMA_BF16(acc[g], aC.v,  bC[g].v);
        WMMA_BF16(acc[g], a2C.v, b2C[g].v);
    }

    // ---- fused LSTM cell: lane owns 8 (batch,h) pairs, col hc, rows b_r ----
    const int hc = hcol0 + m;
    #pragma unroll
    for (int r = 0; r < 8; ++r) {
        const int b_r = row0 + half * 8 + r;
        const size_t off = (size_t)b_r * RD_H + hc;
        const float ig = acc[0][r];
        const float fg = acc[1][r];
        const float gg = acc[2][r];
        const float og = acc[3][r];
        const float c_new = sigm(fg) * c_st[off] + sigm(ig) * tanhf(gg);
        const float h_new = sigm(og) * tanhf(c_new);
        c_st[off]  = c_new;
        h_out[off] = f2bf(h_new);
    }
}

// ---------------------------------------------------------------------------
// Per-row argmax over V (float4 vectorized), then gather E_bf[idx].
// One block per batch row; first-index wins ties.
// ---------------------------------------------------------------------------
__global__ __launch_bounds__(256)
void rd_argmax_gather(const float* __restrict__ pred, int ldo,
                      const unsigned short* __restrict__ E_bf,
                      unsigned short* __restrict__ embed_bf) {
    __shared__ float sv[256];
    __shared__ int   si[256];
    const int b   = blockIdx.x;
    const int tid = threadIdx.x;
    const float4* row4 = (const float4*)(pred + (size_t)b * (size_t)ldo);

    float best = -3.402823466e+38f;
    int   bi   = 0;
    for (int q = tid; q < RD_V / 4; q += 256) {
        const float4 f = row4[q];
        const int v0 = q * 4;
        if (f.x > best || (f.x == best && v0 < bi))     { best = f.x; bi = v0; }
        if (f.y > best || (f.y == best && v0 + 1 < bi)) { best = f.y; bi = v0 + 1; }
        if (f.z > best || (f.z == best && v0 + 2 < bi)) { best = f.z; bi = v0 + 2; }
        if (f.w > best || (f.w == best && v0 + 3 < bi)) { best = f.w; bi = v0 + 3; }
    }
    sv[tid] = best; si[tid] = bi;
    __syncthreads();
    for (int s = 128; s > 0; s >>= 1) {
        if (tid < s) {
            const float ov = sv[tid + s];
            const int   oi = si[tid + s];
            if (ov > sv[tid] || (ov == sv[tid] && oi < si[tid])) {
                sv[tid] = ov; si[tid] = oi;
            }
        }
        __syncthreads();
    }
    const int idx = si[0];
    const uint4* src = (const uint4*)(E_bf + (size_t)idx * RD_H);
    uint4*       dst = (uint4*)(embed_bf + (size_t)b * RD_H);
    for (int i = tid; i < RD_H / 8; i += 256) dst[i] = src[i];
}

// ---------------------------------------------------------------------------
// Utility kernels
// ---------------------------------------------------------------------------
__global__ void rd_cvt_f32_bf16(const float* __restrict__ src,
                                unsigned short* __restrict__ dst, int n4) {
    const int i = blockIdx.x * blockDim.x + threadIdx.x;   // one float4 each
    if (i < n4) {
        const float4 f = ((const float4*)src)[i];
        ushort4 o;
        o.x = f2bf(f.x); o.y = f2bf(f.y); o.z = f2bf(f.z); o.w = f2bf(f.w);
        ((ushort4*)dst)[i] = o;
    }
}

__global__ void rd_zero_state(unsigned short* __restrict__ h_bufs,  // 4*[64,1024]
                              float* __restrict__ c, int n) {       // n = 2*64*1024
    const int i = blockIdx.x * blockDim.x + threadIdx.x;
    if (i < n) {
        h_bufs[i]     = 0;
        h_bufs[n + i] = 0;
        c[i]          = 0.0f;
    }
}

// ---------------------------------------------------------------------------
extern "C" void kernel_launch(void* const* d_in, const int* in_sizes, int n_in,
                              void* d_out, int out_size, void* d_ws, size_t ws_size,
                              hipStream_t stream) {
    (void)in_sizes; (void)n_in; (void)out_size; (void)ws_size;

    const float* x      = (const float*)d_in[0];   // [64,2048]
    const float* proj_W = (const float*)d_in[1];   // [1024,2048]
    const float* proj_b = (const float*)d_in[2];   // [1024]
    const float* W_ih   = (const float*)d_in[3];   // [2,4096,1024]
    const float* W_hh   = (const float*)d_in[4];   // [2,4096,1024]
    const float* b_ih   = (const float*)d_in[5];   // [2,4096]
    const float* b_hh   = (const float*)d_in[6];   // [2,4096]
    const float* E      = (const float*)d_in[7];   // [32000,1024]
    const float* pred_b = (const float*)d_in[8];   // [32000]
    float* out          = (float*)d_out;           // [64,60,32000]

    // ---- workspace carve-up (bf16 region then f32 region) ----
    unsigned short* u = (unsigned short*)d_ws;
    size_t o = 0;
    unsigned short* x_bf     = u + o; o += (size_t)RD_B * RD_IN;
    unsigned short* projW_bf = u + o; o += (size_t)RD_H * RD_IN;
    unsigned short* Wih_bf   = u + o; o += (size_t)2 * RD_G * RD_H;
    unsigned short* Whh_bf   = u + o; o += (size_t)2 * RD_G * RD_H;
    unsigned short* E_bf     = u + o; o += (size_t)RD_V * RD_H;
    unsigned short* embed_bf = u + o; o += (size_t)RD_B * RD_H;
    const size_t HS = (size_t)RD_B * RD_H;         // one h buffer
    unsigned short* h_bufs   = u + o; o += 4 * HS; // l0:ping,pong  l1:ping,pong
    unsigned short* h0[2] = { h_bufs,          h_bufs + HS     };
    unsigned short* h1[2] = { h_bufs + 2 * HS, h_bufs + 3 * HS };
    float* c_st = (float*)(u + o);                 // 2 * [64,1024] f32
    float* c0 = c_st, * c1 = c_st + HS;

    // ---- per-call conversions to bf16 + state zeroing ----
    {
        int n4;
        n4 = RD_B * RD_IN / 4;    rd_cvt_f32_bf16<<<(n4 + 255) / 256, 256, 0, stream>>>(x, x_bf, n4);
        n4 = RD_H * RD_IN / 4;    rd_cvt_f32_bf16<<<(n4 + 255) / 256, 256, 0, stream>>>(proj_W, projW_bf, n4);
        n4 = 2 * RD_G * RD_H / 4; rd_cvt_f32_bf16<<<(n4 + 255) / 256, 256, 0, stream>>>(W_ih, Wih_bf, n4);
        n4 = 2 * RD_G * RD_H / 4; rd_cvt_f32_bf16<<<(n4 + 255) / 256, 256, 0, stream>>>(W_hh, Whh_bf, n4);
        n4 = RD_V * RD_H / 4;     rd_cvt_f32_bf16<<<(n4 + 255) / 256, 256, 0, stream>>>(E, E_bf, n4);
        const int nz = 2 * RD_B * RD_H;
        rd_zero_state<<<(nz + 255) / 256, 256, 0, stream>>>(h_bufs, c_st, nz);
    }

    // ---- initial projection: embed = relu(x @ proj_W^T + proj_b) ----
    rd_gemm<true, false, true><<<RD_H / 128, 256, 0, stream>>>(
        x_bf, projW_bf, proj_b, nullptr, embed_bf, RD_H, RD_IN, RD_H);

    // ---- 60 sequential decode steps ----
    for (int t = 0; t < RD_T; ++t) {
        const int pi = t & 1, po = pi ^ 1;   // ping-pong h buffers

        // layer 0 (fused GEMM + cell): in=embed, recurrent=h0[pi] -> h0[po]
        rd_lstm_layer<<<RD_H / 32, 256, 0, stream>>>(
            embed_bf, h0[pi], Wih_bf, Whh_bf, b_ih, b_hh, c0, h0[po]);

        // layer 1: in = new h0, recurrent = h1[pi] -> h1[po]
        rd_lstm_layer<<<RD_H / 32, 256, 0, stream>>>(
            h0[po], h1[pi],
            Wih_bf + (size_t)RD_G * RD_H, Whh_bf + (size_t)RD_G * RD_H,
            b_ih + RD_G, b_hh + RD_G, c1, h1[po]);

        // vocab projection: pred[:,t,:] = h1_new @ E^T + pred_b
        rd_gemm<false, true, false><<<RD_V / 128, 256, 0, stream>>>(
            h1[po], E_bf, pred_b, out + (size_t)t * RD_V, nullptr,
            RD_V, RD_H, RD_T * RD_V);

        // greedy feedback: embed = E[argmax(pred)]
        rd_argmax_gather<<<RD_B, 256, 0, stream>>>(
            out + (size_t)t * RD_V, RD_T * RD_V, E_bf, embed_bf);
    }
}
